// sig_add_21629455303047
// MI455X (gfx1250) — matrix operand order repaired and checked by reading, not compiled, not run on hardware
//
#include <hip/hip_runtime.h>
#include <hip/hip_bf16.h>
#include <stdint.h>
#include <stddef.h>

typedef _Float16 h16;
typedef __attribute__((ext_vector_type(16))) _Float16 v16h;
typedef __attribute__((ext_vector_type(8)))  float    v8f;
typedef __attribute__((ext_vector_type(4)))  unsigned int u32x4;
typedef __attribute__((ext_vector_type(8)))  int          i32x8;
typedef __attribute__((ext_vector_type(4)))  int          i32x4;

#define BN_EPS 1e-5f

// ---------------------------------------------------------------------------
// Small utility kernels
// ---------------------------------------------------------------------------
__global__ void k_bn_fold(const float* __restrict__ gamma, const float* __restrict__ beta,
                          const float* __restrict__ mean,  const float* __restrict__ var,
                          int C, float* __restrict__ scale, float* __restrict__ bias) {
  int i = blockIdx.x * blockDim.x + threadIdx.x;
  if (i < C) {
    float s  = gamma[i] * rsqrtf(var[i] + BN_EPS);
    scale[i] = s;
    bias[i]  = beta[i] - mean[i] * s;
  }
}

__global__ void k_f32_to_f16(const float* __restrict__ src, h16* __restrict__ dst, int n) {
  for (int i = blockIdx.x * blockDim.x + threadIdx.x; i < n; i += gridDim.x * blockDim.x)
    dst[i] = (h16)src[i];
}

// f32 OIHW weights -> f16 [Cout][Kpad] with zero padding (Kpad = roundup(K,32)).
__global__ void k_cvt_pad(const float* __restrict__ src, h16* __restrict__ dst,
                          int Cout, int K, int Kpad) {
  for (int i = blockIdx.x * blockDim.x + threadIdx.x; i < Cout * Kpad;
       i += gridDim.x * blockDim.x) {
    int m = i / Kpad, k = i - m * Kpad;
    dst[i] = (k < K) ? (h16)src[(size_t)m * K + k] : (h16)0.f;
  }
}

// ---------------------------------------------------------------------------
// Implicit-GEMM convolution with WMMA f16->f32, fused BN (+residual) (+ReLU).
//   GEMM: M = Cout, N = Nb*OH*OW, K = Cin*KH*KW (padded to Kpad)
//   Block: 256 threads = 8 waves (2x4). Block tile 128x128, BK = 32.
//   Double-buffered pipeline: Tensor Data Mover DMAs the next weight tile and
//   threads stage the next im2col tile while WMMAs consume the current one.
// ---------------------------------------------------------------------------
__global__ __launch_bounds__(256) void k_conv_wmma(
    const h16* __restrict__ Wt, const h16* __restrict__ X, h16* __restrict__ Y,
    const float* __restrict__ scale, const float* __restrict__ bias,
    const h16* __restrict__ res, int relu,
    int Nb, int Cin, int H, int Wd, int Cout, int OH, int OW,
    int KH, int KW, int stride, int pad, int Kpad)
{
  __shared__ __align__(128) h16 As[2][128 * 32];   // [m][k], k-contiguous
  __shared__ __align__(128) h16 Bs[2][128 * 32];   // [n][k], k-contiguous

  const int K    = Cin * KH * KW;
  const int khw  = KH * KW;
  const int HW   = H * Wd;
  const int OHOW = OH * OW;
  const int Npix = Nb * OHOW;

  const int tid  = threadIdx.x;
  const int lane = tid & 31;
  const int wave = tid >> 5;
  const int wm   = wave >> 2;    // 0..1  (M direction)
  const int wn   = wave & 3;     // 0..3  (N direction)

  const int mBlock = blockIdx.y * 128;
  const int nBlock = blockIdx.x * 128;

  // Staging: each thread fills one 16-half chunk of one row of Bs.
  const int sRow = tid >> 1;          // 0..127
  const int sK   = (tid & 1) * 16;    // 0 or 16

  const int gn = nBlock + sRow;       // flattened output pixel
  int  bb = 0, ohB = 0, owB = 0;
  bool pixOk = (gn < Npix);
  if (pixOk) {
    bb = gn / OHOW;
    int r  = gn - bb * OHOW;
    int oh = r / OW;
    int ow = r - oh * OW;
    ohB = oh * stride - pad;
    owB = ow * stride - pad;
  }
  const size_t pixBase = (size_t)bb * Cin * HW;            // batch base in X
  const size_t pix1    = pixBase + (size_t)ohB * Wd + owB; // 1x1 fast-path base

  v8f acc[4][2];
#pragma unroll
  for (int mt = 0; mt < 4; ++mt)
#pragma unroll
    for (int nt = 0; nt < 2; ++nt)
#pragma unroll
      for (int r = 0; r < 8; ++r) acc[mt][nt][r] = 0.f;

  // Invariant pieces of the Tensor DMA descriptor (wave-uniform).
  const int rowsLeft = (Cout - mBlock) < 128 ? (Cout - mBlock) : 128;

  // ---- TDM issue for A tile kt2 into As[buf] (wave 0 only) ----
  auto issueA = [&](int kt2, int buf) {
    if (wave == 0) {
      const int ktBase = kt2 << 5;
      uint64_t ga = (uint64_t)(uintptr_t)Wt +
                    (((uint64_t)mBlock * Kpad + (uint64_t)ktBase) << 1);
      int colsLeft = Kpad - ktBase;      // multiple of 32, >= 32
      u32x4 g0;
      g0[0] = 1u;                                            // count=1 (user D#)
      g0[1] = (uint32_t)(uintptr_t)(&As[buf][0]);            // lds_addr
      g0[2] = (uint32_t)ga;                                  // global_addr[31:0]
      g0[3] = (uint32_t)((ga >> 32) & 0x01FFFFFFu) | (2u << 30); // addr[56:32], type=2
      i32x8 g1;
      g1[0] = (int)(1u << 16);                               // data_size=1 (2 bytes)
      g1[1] = (int)(((uint32_t)colsLeft & 0xFFFFu) << 16);   // tensor_dim0[15:0]
      g1[2] = (int)(((uint32_t)colsLeft >> 16) |
                    (((uint32_t)rowsLeft & 0xFFFFu) << 16)); // dim0[31:16], dim1[15:0]
      g1[3] = (int)(((uint32_t)rowsLeft >> 16) | (32u << 16)); // dim1[31:16], tile_dim0=32
      g1[4] = 128;                                           // tile_dim1=128, tile_dim2=0
      g1[5] = (int)(uint32_t)Kpad;                           // tensor_dim0_stride[31:0]
      g1[6] = 0;                                             // stride hi, dim1_stride lo
      g1[7] = 0;
      i32x4 gz4 = {0, 0, 0, 0};
      i32x8 gz8 = {0, 0, 0, 0, 0, 0, 0, 0};
      __builtin_amdgcn_tensor_load_to_lds(g0, g1, gz4, gz4, gz8, 0);
    }
  };

  // ---- im2col staging of B tile kt2 into Bs[buf] (all threads) ----
  auto stageB = [&](int kt2, int buf) {
    int k0 = (kt2 << 5) + sK;
    if (khw == 1) {
      // 1x1 conv (pad==0): addr = pix1 + k*HW, no divides.
      if (pixOk && k0 + 32 < K)                    // speculative next-tile prefetch
        __builtin_prefetch(&X[pix1 + (size_t)(k0 + 32) * HW], 0, 1);
#pragma unroll 4
      for (int i = 0; i < 16; ++i) {
        int k = k0 + i;
        h16 v = (h16)0.f;
        if (pixOk && k < K) v = X[pix1 + (size_t)k * HW];
        Bs[buf][sRow * 32 + sK + i] = v;
      }
    } else {
      // General conv: decode (cin,kh,kw) once, then increment.
      int cin = k0 / khw;
      int rk  = k0 - cin * khw;
      int kh  = rk / KW;
      int kw  = rk - kh * KW;
#pragma unroll 4
      for (int i = 0; i < 16; ++i) {
        h16 v = (h16)0.f;
        if (pixOk && (k0 + i) < K) {
          int ih = ohB + kh, iw = owB + kw;
          if ((unsigned)ih < (unsigned)H && (unsigned)iw < (unsigned)Wd)
            v = X[pixBase + (size_t)cin * HW + (size_t)ih * Wd + iw];
        }
        Bs[buf][sRow * 32 + sK + i] = v;
        if (++kw == KW) { kw = 0; if (++kh == KH) { kh = 0; ++cin; } }
      }
    }
  };

  // ---- fragment loads (CDNA5 16-bit A/B VGPR layouts) + 8 WMMAs ----
  auto compute = [&](int buf) {
    v16h afr[4], bfr[2];
    const int am   = lane & 15;
    const int aksl = (lane & 16) ? 8 : 0;     // lanes>=16 hold K+8 / K+24 groups
#pragma unroll
    for (int mt = 0; mt < 4; ++mt) {
      const h16* p = &As[buf][(wm * 64 + mt * 16 + am) * 32 + aksl];
#pragma unroll
      for (int i = 0; i < 8; ++i) { afr[mt][i] = p[i]; afr[mt][8 + i] = p[16 + i]; }
    }
    const int bn_  = lane & 15;
    const int bksl = (lane & 16) ? 8 : 0;     // lanes>=16 hold K+8 / K+24 groups
#pragma unroll
    for (int nt = 0; nt < 2; ++nt) {
      const h16* p = &Bs[buf][(wn * 32 + nt * 16 + bn_) * 32 + bksl];
#pragma unroll
      for (int i = 0; i < 8; ++i) { bfr[nt][i] = p[i]; bfr[nt][8 + i] = p[16 + i]; }
    }
#pragma unroll
    for (int mt = 0; mt < 4; ++mt)
#pragma unroll
      for (int nt = 0; nt < 2; ++nt)
        acc[mt][nt] = __builtin_amdgcn_wmma_f32_16x16x32_f16(
            false, afr[mt], false, bfr[nt], (short)0, acc[mt][nt], false, false);
  };

  // ---- software pipeline: overlap TDM + im2col loads with WMMA compute ----
  const int kTiles = Kpad >> 5;
  issueA(0, 0);
  stageB(0, 0);
  if (wave == 0) __builtin_amdgcn_s_wait_tensorcnt(0);
  __syncthreads();
  for (int kt = 0; kt < kTiles; ++kt) {
    const int p = kt & 1;
    if (kt + 1 < kTiles) {              // stage next tile into the other buffer
      issueA(kt + 1, p ^ 1);
      stageB(kt + 1, p ^ 1);
    }
    compute(p);                         // WMMAs on current buffer (concurrent w/ TDM)
    if (wave == 0 && kt + 1 < kTiles) __builtin_amdgcn_s_wait_tensorcnt(0);
    __syncthreads();                    // retire tile: reads of p done, p^1 landed
  }

  // ---- fused epilogue: BN scale/bias, optional residual, optional ReLU ----
  const int nCol = lane & 15;
  const int mHi  = (lane & 16) ? 8 : 0;       // C/D layout: VGPR r -> M=r (lo) / r+8 (hi)
#pragma unroll
  for (int nt = 0; nt < 2; ++nt) {
    int n = nBlock + wn * 32 + nt * 16 + nCol;
    if (n < Npix) {
      int b2 = n / OHOW;
      int r2 = n - b2 * OHOW;
      size_t pixOff = (size_t)b2 * Cout * OHOW + r2;   // + m*OHOW gives NCHW index
#pragma unroll
      for (int mt = 0; mt < 4; ++mt)
#pragma unroll
        for (int r = 0; r < 8; ++r) {
          int m = mBlock + wm * 64 + mt * 16 + mHi + r;
          if (m < Cout) {
            float v = acc[mt][nt][r] * scale[m] + bias[m];
            size_t idx = pixOff + (size_t)m * OHOW;
            if (res)  v += (float)res[idx];
            if (relu) v = v > 0.f ? v : 0.f;
            Y[idx] = (h16)v;
          }
        }
    }
  }
}

// ---------------------------------------------------------------------------
// Pooling / FC / correction kernels
// ---------------------------------------------------------------------------
__global__ void k_maxpool(const h16* __restrict__ in, h16* __restrict__ out,
                          int Nb, int C, int H, int W, int OH, int OW) {
  int idx   = blockIdx.x * blockDim.x + threadIdx.x;
  int total = Nb * C * OH * OW;
  if (idx >= total) return;
  int ow = idx % OW; int t = idx / OW;
  int oh = t % OH;   t /= OH;
  int c  = t % C;    int b = t / C;
  float m = -3.0e38f;
  for (int kh = 0; kh < 3; ++kh)
    for (int kw = 0; kw < 3; ++kw) {
      int ih = oh * 2 - 1 + kh, iw = ow * 2 - 1 + kw;
      if ((unsigned)ih < (unsigned)H && (unsigned)iw < (unsigned)W) {
        float v = (float)in[(((size_t)b * C + c) * H + ih) * W + iw];
        m = v > m ? v : m;
      }
    }
  out[idx] = (h16)m;
}

__global__ void k_avgpool(const h16* __restrict__ in, float* __restrict__ out,
                          int Nb, int C, int H, int W) {
  int idx = blockIdx.x * blockDim.x + threadIdx.x;
  if (idx >= Nb * C) return;
  const h16* p = &in[(size_t)idx * H * W];
  float s = 0.f;
  for (int i = 0; i < H * W; ++i) s += (float)p[i];
  out[idx] = s / (float)(H * W);
}

__global__ void k_fc(const float* __restrict__ pooled, const float* __restrict__ fw,
                     const float* __restrict__ fb, float* __restrict__ out,
                     int B, int Kd, int NC) {
  int idx = blockIdx.x * blockDim.x + threadIdx.x;
  if (idx >= B * NC) return;
  int b = idx / NC, c = idx - b * NC;
  const float* pr = &pooled[(size_t)b * Kd];
  const float* wr = &fw[(size_t)c * Kd];
  float s = fb[c];
  for (int k = 0; k < Kd; ++k) s += pr[k] * wr[k];
  out[idx] = s;
}

// shift = out[:,0::2], inv = 1/out[:,1::2]; invsum[b] = sum_c inv; s = sum(shift*inv)
__global__ void k_corr_stats(const float* __restrict__ logits, int NC,
                             float* __restrict__ invsum, float* __restrict__ sOut) {
  __shared__ float sb[8];
  int b = threadIdx.x;
  if (b < 8) {
    float is = 0.f, sv = 0.f;
    for (int c = 0; c < NC / 2; ++c) {
      float inv = 1.0f / logits[b * NC + 2 * c + 1];
      is += inv;
      sv += logits[b * NC + 2 * c] * inv;
    }
    invsum[b] = is;
    sb[b] = sv;
  }
  __syncthreads();
  if (b == 0) { float s = 0.f; for (int k = 0; k < 8; ++k) s += sb[k]; *sOut = s; }
}

// corr[chw] = sum_b invsum[b] * x[b,chw] - s
__global__ void k_corr(const float* __restrict__ x, const float* __restrict__ invsum,
                       const float* __restrict__ sval, float* __restrict__ corr, int CHW) {
  int idx = blockIdx.x * blockDim.x + threadIdx.x;
  if (idx >= CHW) return;
  float c = -(*sval);
  for (int b = 0; b < 8; ++b) c += invsum[b] * x[(size_t)b * CHW + idx];
  corr[idx] = c;
}

__global__ void k_img(const float* __restrict__ x, const float* __restrict__ corr,
                      h16* __restrict__ img, int N, int CHW) {
  for (int i = blockIdx.x * blockDim.x + threadIdx.x; i < N; i += gridDim.x * blockDim.x)
    img[i] = (h16)(x[i] + corr[i % CHW]);
}

// ---------------------------------------------------------------------------
// Host-side parameter bookkeeping (sorted-dict pytree leaf order)
// ---------------------------------------------------------------------------
struct BNp { const float *beta, *gamma, *mean, *var; };
struct Blk { BNp bn1, bn2, bn3; const float *c1, *c2, *c3;
             BNp dbn; const float* dc; int down; int cin, cmid, stride; };
struct Net { const float *fcb, *fcw; Blk blk[16]; int nblk; BNp stem; const float* stemw; };

static int parse_net(void* const* d_in, int start, Net& net) {
  int i = start;
  auto F   = [&]() { return (const float*)d_in[i++]; };
  auto RBN = [&](BNp& p) { p.beta = F(); p.gamma = F(); p.mean = F(); p.var = F(); };
  net.fcb = F(); net.fcw = F();
  const int cmids[4] = {64, 128, 256, 512};
  const int nb[4]    = {3, 4, 6, 3};
  int cin = 64, bi = 0;
  for (int L = 0; L < 4; ++L)
    for (int b = 0; b < nb[L]; ++b) {
      Blk& B = net.blk[bi++];
      RBN(B.bn1); RBN(B.bn2); RBN(B.bn3);
      B.c1 = F(); B.c2 = F(); B.c3 = F();
      B.down = (b == 0);
      if (B.down) { RBN(B.dbn); B.dc = F(); } else { B.dc = nullptr; }
      B.cin = cin; B.cmid = cmids[L];
      B.stride = (L > 0 && b == 0) ? 2 : 1;
      cin = cmids[L] * 4;
    }
  net.nblk = bi;
  RBN(net.stem);
  net.stemw = F();
  return i;
}

static void conv(hipStream_t s, const float* w32, const BNp& bn,
                 const h16* in, h16* out, const h16* res, int relu,
                 int Nb, int Cin, int H, int W, int Cout, int KH, int KW,
                 int stride, int pad,
                 h16* wbuf, float* scale, float* bias, int& OH, int& OW) {
  int K    = Cin * KH * KW;
  int Kpad = (K + 31) & ~31;
  k_bn_fold<<<(Cout + 255) / 256, 256, 0, s>>>(bn.gamma, bn.beta, bn.mean, bn.var, Cout, scale, bias);
  k_cvt_pad<<<(Cout * Kpad + 255) / 256, 256, 0, s>>>(w32, wbuf, Cout, K, Kpad);
  OH = (H + 2 * pad - KH) / stride + 1;
  OW = (W + 2 * pad - KW) / stride + 1;
  int Npix = Nb * OH * OW;
  dim3 g((Npix + 127) / 128, (Cout + 127) / 128);
  k_conv_wmma<<<g, 256, 0, s>>>(wbuf, in, out, scale, bias, res, relu,
                                Nb, Cin, H, W, Cout, OH, OW, KH, KW, stride, pad, Kpad);
}

static void run_net(hipStream_t s, const Net& net, const h16* xin, h16* bufs[4],
                    h16* wbuf, float* scale, float* bias, float* pooled,
                    float* logits, int nclass) {
  int OH, OW;
  // stem 7x7/2 pad 3 + BN + ReLU
  conv(s, net.stemw, net.stem, xin, bufs[0], nullptr, 1,
       8, 3, 224, 224, 64, 7, 7, 2, 3, wbuf, scale, bias, OH, OW);
  // maxpool 3x3/2 pad 1: 112 -> 56
  {
    int n = 8 * 64 * 56 * 56;
    k_maxpool<<<(n + 255) / 256, 256, 0, s>>>(bufs[0], bufs[1], 8, 64, 112, 112, 56, 56);
  }
  int cur = 1, H = 56;
  for (int b = 0; b < net.nblk; ++b) {
    const Blk& B = net.blk[b];
    int t1 = (cur + 1) & 3, t2 = (cur + 2) & 3, t3 = (cur + 3) & 3;
    int cout = B.cmid * 4;
    conv(s, B.c1, B.bn1, bufs[cur], bufs[t1], nullptr, 1,
         8, B.cin, H, H, B.cmid, 1, 1, 1, 0, wbuf, scale, bias, OH, OW);
    conv(s, B.c2, B.bn2, bufs[t1], bufs[t2], nullptr, 1,
         8, B.cmid, H, H, B.cmid, 3, 3, B.stride, 1, wbuf, scale, bias, OH, OW);
    int Ho = OH;
    const h16* sc;
    if (B.down) {
      conv(s, B.dc, B.dbn, bufs[cur], bufs[t3], nullptr, 0,
           8, B.cin, H, H, cout, 1, 1, B.stride, 0, wbuf, scale, bias, OH, OW);
      sc = bufs[t3];
    } else {
      sc = bufs[cur];
    }
    conv(s, B.c3, B.bn3, bufs[t2], bufs[t1], sc, 1,
         8, B.cmid, Ho, Ho, cout, 1, 1, 1, 0, wbuf, scale, bias, OH, OW);
    cur = t1;
    H = Ho;
  }
  k_avgpool<<<(8 * 2048 + 255) / 256, 256, 0, s>>>(bufs[cur], pooled, 8, 2048, 7, 7);
  k_fc<<<1, 256, 0, s>>>(pooled, net.fcw, net.fcb, logits, 8, 2048, nclass);
}

// ---------------------------------------------------------------------------
// Entry point
// ---------------------------------------------------------------------------
extern "C" void kernel_launch(void* const* d_in, const int* in_sizes, int n_in,
                              void* d_out, int out_size, void* d_ws, size_t ws_size,
                              hipStream_t stream) {
  (void)in_sizes; (void)n_in; (void)out_size; (void)ws_size;
  const float* x = (const float*)d_in[0];
  Net base, cls;
  int i = parse_net(d_in, 1, base);
  parse_net(d_in, i, cls);

  // bump allocator in workspace
  char* p = (char*)d_ws;
  auto alloc = [&](size_t bytes) -> char* {
    char* r = p;
    p += (bytes + 255) & ~(size_t)255;
    return r;
  };
  const size_t CHW = (size_t)3 * 224 * 224;
  const size_t XN  = 8 * CHW;
  const size_t BUF = (size_t)8 * 256 * 56 * 56;   // max activation elements

  h16* xf16 = (h16*)alloc(XN * 2);
  h16* bufs[4];
  for (int k = 0; k < 4; ++k) bufs[k] = (h16*)alloc(BUF * 2);
  h16*   wbuf   = (h16*)alloc((size_t)512 * 512 * 3 * 3 * 2);  // largest conv weight (Kpad-safe)
  float* scale  = (float*)alloc(2048 * 4);
  float* bias   = (float*)alloc(2048 * 4);
  float* pooled = (float*)alloc(8 * 2048 * 4);
  float* logits = (float*)alloc(8 * 20 * 4);
  float* invsum = (float*)alloc(8 * 4);
  float* sval   = (float*)alloc(4);
  float* corr   = (float*)alloc(CHW * 4);

  // first network (base_params, 20 classes)
  k_f32_to_f16<<<1024, 256, 0, stream>>>(x, xf16, (int)XN);
  run_net(stream, base, xf16, bufs, wbuf, scale, bias, pooled, logits, 20);

  // correction: img = x + (sum_b invsum[b]*x[b] - sum(shift*inv))
  k_corr_stats<<<1, 32, 0, stream>>>(logits, 20, invsum, sval);
  k_corr<<<(int)((CHW + 255) / 256), 256, 0, stream>>>(x, invsum, sval, corr, (int)CHW);
  k_img<<<1024, 256, 0, stream>>>(x, corr, xf16, (int)XN, (int)CHW);

  // second network (cls_params, 2 classes) -> d_out (8x2 f32)
  run_net(stream, cls, xf16, bufs, wbuf, scale, bias, pooled, (float*)d_out, 2);
}
